// OptimizedTwoSidedVLUTConv2d_29025388986485
// MI455X (gfx1250) — matrix operand order repaired and checked by reading, not compiled, not run on hardware
//
#include <hip/hip_runtime.h>
#include <stdint.h>

// ---- CDNA5 wave32 WMMA types ----
typedef __attribute__((ext_vector_type(16))) _Float16 v16h;
typedef __attribute__((ext_vector_type(8)))  _Float16 v8h;
typedef __attribute__((ext_vector_type(8)))  float    v8f;

// Problem constants
#define BATCH 16
#define CIN   128
#define HW    56
#define HWP   58                 // padded spatial (1-px halo)
#define COUT  256
#define LPOS  (HW*HW)            // 3136
#define KDIM  (CIN*9)            // 1152
#define MTOT  (BATCH*LPOS)       // 50176

// Tiling
#define BM 128
#define BN 128
#define BK 32
#define LDSS 40                  // halves per LDS row (80B, 16B aligned)
#define NSTEP 36                 // 9 taps x 4 c-chunks

// Workspace layout: [W-pack f16][padded NHWC f16 image]
#define WS_W_HALVES ((COUT/BN)*NSTEP*BN*BK)          // 294912
#define WS_W_BYTES  ((size_t)WS_W_HALVES*2)          // 589824
#define XP_HALVES   ((size_t)BATCH*HWP*HWP*CIN)      // 6,889,472
#define XP_BYTES    (XP_HALVES*2)
#define WS_TOTAL    (WS_W_BYTES + XP_BYTES)

union FragH { v16h v; v8h h[2]; };
template <int V> struct IC { static constexpr int value = V; };

// ---- CDNA5 async global->LDS copy (ASYNCcnt), inline asm ----
// ISA 08 §4.4: the imm offset is added to the LDS and the global address alike  -> the same
// immediate steps BOTH addresses; we compensate the LDS side with a u32 sub.
template <int IMM>
__device__ __forceinline__ void async_b128(uint32_t lds_comp, const void* gbase) {
    asm volatile("global_load_async_to_lds_b128 %0, %1, off offset:%2"
                 :: "v"(lds_comp), "v"(gbase), "i"(IMM) : "memory");
}
__device__ __forceinline__ void wait_async0() {
    asm volatile("s_wait_asynccnt 0x0" ::: "memory");
}

// ---------------- pre-pass 1: weights f32 (cout, c*9+off) -> f16 K-permuted tiles ----------------
__global__ __launch_bounds__(256, 1)
void prepack_w(const float* __restrict__ wgt, _Float16* __restrict__ ws)
{
    const int tile = blockIdx.x;            // nb*36 + s
    const int nb = tile / NSTEP;
    const int s  = tile - nb * NSTEP;
    const int off = s >> 2;
    const int c0  = (s & 3) * BK;
    #pragma unroll
    for (int i = threadIdx.x; i < BN * BK; i += 256) {
        const int nl = i >> 5;
        const int kl = i & 31;
        ws[(size_t)tile * (BN * BK) + i] =
            (_Float16)wgt[(size_t)(nb * BN + nl) * KDIM + (c0 + kl) * 9 + off];
    }
}

// ---------------- pre-pass 2: NCHW f32 -> halo-padded NHWC f16 ----------------
__global__ __launch_bounds__(256, 1)
void transpose_in(const float* __restrict__ in, _Float16* __restrict__ xp)
{
    const int bh = blockIdx.x;              // 0..BATCH*HW-1
    const int b  = bh / HW;
    const int h  = bh - b * HW;
    #pragma unroll
    for (int e = threadIdx.x; e < HW * CIN; e += 256) {
        const int c = e & 127;
        const int w = e >> 7;
        const float v = in[((size_t)(b * CIN + c)) * LPOS + h * HW + w];
        xp[((size_t)(b * HWP + h + 1) * HWP + (w + 1)) * CIN + c] = (_Float16)v;
    }
}

// step-S displacement (bytes) of the X async load within the padded NHWC image
__host__ __device__ constexpr int x_disp(int S) {
    return (((S >> 2) / 3) * HWP + ((S >> 2) % 3)) * CIN * 2 + (S & 3) * BK * 2;
}
__host__ __device__ constexpr int w_disp(int S) { return S * BN * BK * 2; }

// ---------------- main GEMM ----------------
template <bool USE_WS>
__global__ __launch_bounds__(256, 1)
void vlut_conv2d_wmma_f16(const float* __restrict__ in,
                          const float* __restrict__ wgt,
                          const float* __restrict__ bias,
                          const _Float16* __restrict__ wsw,   // packed weights
                          const _Float16* __restrict__ wsx,   // padded NHWC image
                          float* __restrict__ out)
{
    __shared__ _Float16 ldsX[2][BM * LDSS];
    __shared__ _Float16 ldsW[2][BN * LDSS];

    const int t      = threadIdx.x;
    const int lane   = t & 31;
    const int wave   = t >> 5;
    const int wave_n = wave & 1;
    const int wave_m = wave >> 1;

    const int m0 = blockIdx.x * BM;
    const int n0 = blockIdx.y * BN;

    // per-thread im2col geometry (fixed)
    const int mrow  = t & 127;
    const int khalf = t >> 7;
    const int mg0  = m0 + mrow;
    const int bb   = mg0 / LPOS;
    const int pos  = mg0 - bb * LPOS;
    const int oh   = pos / HW;
    const int ow   = pos - oh * HW;

    // single 64-bit global bases for the whole kernel (stepping via imm offsets)
    const _Float16* xbase = wsx + ((size_t)(bb * HWP + oh) * HWP + ow) * CIN + khalf * 16;
    const _Float16* wbase = USE_WS ? (wsw + (size_t)blockIdx.y * NSTEP * (BN * BK) + t * 16)
                                   : nullptr;
    const uint32_t ldsx_d[2] = {
        (uint32_t)(uintptr_t)&ldsX[0][mrow * LDSS + khalf * 16],
        (uint32_t)(uintptr_t)&ldsX[1][mrow * LDSS + khalf * 16] };
    const uint32_t ldsw_d[2] = {
        (uint32_t)(uintptr_t)&ldsW[0][(t >> 1) * LDSS + (t & 1) * 16],
        (uint32_t)(uintptr_t)&ldsW[1][(t >> 1) * LDSS + (t & 1) * 16] };

    // fallback base
    const float* inb = in + (size_t)bb * (CIN * LPOS);

    v8f acc[2][4];
    {
        v8f z = {};
        #pragma unroll
        for (int mi = 0; mi < 2; ++mi)
            #pragma unroll
            for (int ni = 0; ni < 4; ++ni) acc[mi][ni] = z;
    }

    const int arow = lane & 15;
    const int ahi  = lane >> 4;

    // ---------- async issue for step S (immediates are constexpr) ----------
    auto issue = [&](auto sc) {
        constexpr int S    = decltype(sc)::value;
        constexpr int XOFF = x_disp(S);
        constexpr int WOFF = w_disp(S);
        constexpr int buf  = S & 1;
        const uint32_t dx = ldsx_d[buf] - (uint32_t)XOFF;   // u32 wrap cancels imm
        const uint32_t dw = ldsw_d[buf] - (uint32_t)WOFF;
        async_b128<XOFF +  0>(dx, xbase);
        async_b128<XOFF + 16>(dx, xbase);
        async_b128<WOFF +  0>(dw, wbase);
        async_b128<WOFF + 16>(dw, wbase);
    };

    // ---------- WMMA compute ----------
    auto compute = [&](int buf) {
        FragH a[2], b[4];
        #pragma unroll
        for (int mi = 0; mi < 2; ++mi) {
            const _Float16* p = &ldsX[buf][(wave_m * 32 + mi * 16 + arow) * LDSS + ahi * 8];
            a[mi].h[0] = *(const v8h*)(p);
            a[mi].h[1] = *(const v8h*)(p + 16);
        }
        #pragma unroll
        for (int ni = 0; ni < 4; ++ni) {
            const _Float16* p = &ldsW[buf][(wave_n * 64 + ni * 16 + arow) * LDSS + ahi * 8];
            b[ni].h[0] = *(const v8h*)(p);
            b[ni].h[1] = *(const v8h*)(p + 16);
        }
        #pragma unroll
        for (int mi = 0; mi < 2; ++mi)
            #pragma unroll
            for (int ni = 0; ni < 4; ++ni)
                acc[mi][ni] = __builtin_amdgcn_wmma_f32_16x16x32_f16(
                    false, a[mi].v, false, b[ni].v,
                    (short)0, acc[mi][ni], false, false);
    };

    if constexpr (USE_WS) {
        // template-recursive K loop: issue(S+1) overlaps compute(S)
        issue(IC<0>{});
        wait_async0();
        __syncthreads();
        kloop(IC<0>{}, issue, compute);
    } else {
        // fallback: synchronous f32 gather (no workspace)
        auto load_sync = [&](int buf, int s) {
            const int off = s >> 2;
            const int c0s = (s & 3) * BK;
            const int kh  = off / 3;
            const int kw  = off - kh * 3;
            const int ih  = oh - 1 + kh;
            const int iw  = ow - 1 + kw;
            const bool valid = ((unsigned)ih < (unsigned)HW) & ((unsigned)iw < (unsigned)HW);
            const int ihc = min(max(ih, 0), HW - 1);
            const int iwc = min(max(iw, 0), HW - 1);
            const float* bx = inb + (size_t)(c0s + khalf * 16) * LPOS + ihc * HW + iwc;
            float v[16];
            #pragma unroll
            for (int j = 0; j < 16; ++j) v[j] = bx[j * LPOS];
            #pragma unroll
            for (int j = 0; j < 16; ++j) v[j] = valid ? v[j] : 0.0f;
            v8h p0, p1;
            #pragma unroll
            for (int j = 0; j < 8; ++j) { p0[j] = (_Float16)v[j]; p1[j] = (_Float16)v[j + 8]; }
            _Float16* dx = &ldsX[buf][mrow * LDSS + khalf * 16];
            *(v8h*)dx = p0;
            *(v8h*)(dx + 8) = p1;
            #pragma unroll
            for (int e = t; e < BN * BK; e += 256) {
                const int kl = e & 31;
                const int nl = e >> 5;
                ldsW[buf][nl * LDSS + kl] =
                    (_Float16)wgt[(size_t)(n0 + nl) * KDIM + (c0s + kl) * 9 + off];
            }
        };
        load_sync(0, 0);
        __syncthreads();
        for (int s = 0; s < NSTEP; ++s) {
            const int cur = s & 1;
            if (s + 1 < NSTEP) load_sync(cur ^ 1, s + 1);
            compute(cur);
            __syncthreads();
        }
    }

    // ---------- epilogue: bias + vectorized stores ----------
    float bv[4];
    #pragma unroll
    for (int ni = 0; ni < 4; ++ni) bv[ni] = bias[n0 + wave_n * 64 + ni * 16 + arow];

    #pragma unroll
    for (int mi = 0; mi < 2; ++mi) {
        const int mg = m0 + wave_m * 32 + mi * 16 + 8 * ahi;   // 8-aligned run
        const int bo = mg / LPOS;
        const int pg = mg - bo * LPOS;
        #pragma unroll
        for (int ni = 0; ni < 4; ++ni) {
            const int n = n0 + wave_n * 64 + ni * 16 + arow;
            float* po = out + ((size_t)(bo * COUT + n)) * LPOS + pg;
            float4 lo = { acc[mi][ni][0] + bv[ni], acc[mi][ni][1] + bv[ni],
                          acc[mi][ni][2] + bv[ni], acc[mi][ni][3] + bv[ni] };
            float4 hi = { acc[mi][ni][4] + bv[ni], acc[mi][ni][5] + bv[ni],
                          acc[mi][ni][6] + bv[ni], acc[mi][ni][7] + bv[ni] };
            *(float4*)(po)     = lo;
            *(float4*)(po + 4) = hi;
        }
    }
}

// template-recursive unrolled K loop (declared after use via forward helper is
// awkward in a kernel, so define as free function template used above)
template <int S, class FIssue, class FCompute>
__device__ __forceinline__ void kloop(IC<S>, FIssue issue, FCompute compute) {
    if constexpr (S < NSTEP) {
        if constexpr (S + 1 < NSTEP) issue(IC<S + 1>{});
        compute(S & 1);
        wait_async0();
        __syncthreads();
        kloop(IC<S + 1>{}, issue, compute);
    }
}

extern "C" void kernel_launch(void* const* d_in, const int* in_sizes, int n_in,
                              void* d_out, int out_size, void* d_ws, size_t ws_size,
                              hipStream_t stream) {
    const float* in   = (const float*)d_in[0];
    const float* wgt  = (const float*)d_in[1];
    const float* bias = (const float*)d_in[2];
    float* out = (float*)d_out;

    dim3 grid(MTOT / BM, COUT / BN, 1);   // (392, 2)
    dim3 block(256, 1, 1);

    if (ws_size >= WS_TOTAL) {
        _Float16* wsw = (_Float16*)d_ws;
        _Float16* wsx = (_Float16*)((char*)d_ws + WS_W_BYTES);
        hipMemsetAsync(wsx, 0, XP_BYTES, stream);
        prepack_w<<<dim3((COUT / BN) * NSTEP), block, 0, stream>>>(wgt, wsw);
        transpose_in<<<dim3(BATCH * HW), block, 0, stream>>>(in, wsx);
        vlut_conv2d_wmma_f16<true><<<grid, block, 0, stream>>>(in, wgt, bias, wsw, wsx, out);
    } else {
        vlut_conv2d_wmma_f16<false><<<grid, block, 0, stream>>>(in, wgt, bias, nullptr, nullptr, out);
    }
}
